// GraphMultiHeadAttention_71236327571572
// MI455X (gfx1250) — compile-verified, matrix operand order checked by reading, not run on hardware
//
#include <hip/hip_runtime.h>

#define B_    4
#define NSEQ  1024
#define DIM   1024
#define HEADS 16
#define HDIM  64

typedef __attribute__((ext_vector_type(16))) __bf16 v16bf;
typedef __attribute__((ext_vector_type(8)))  float  v8f;
typedef __attribute__((ext_vector_type(4)))  unsigned int u32x4;
typedef __attribute__((ext_vector_type(8)))  unsigned int u32x8;

// D = A(16x32 bf16) * B(32x16 bf16) + C(16x16 f32)
__device__ __forceinline__ v8f wmma_bf16(v16bf a, v16bf b, v8f c) {
  return __builtin_amdgcn_wmma_f32_16x16x32_bf16(false, a, false, b, (short)0, c,
                                                 false, false);
}

__device__ __forceinline__ uint32_t lds_addr(const void* p) {
  return (uint32_t)(uintptr_t)p;  // LDS aperture: low 32 bits = LDS byte offset
}

// Async copy 16B global -> LDS (per lane), tracked by ASYNCcnt.
__device__ __forceinline__ void async_load_b128(uint32_t lds_off, const void* gptr) {
  asm volatile("global_load_async_to_lds_b128 %0, %1, off"
               :: "v"(lds_off), "v"(gptr)
               : "memory");
}

// Tensor Data Mover: DMA a 2D tile (tile_d0 x tile_d1, 2-byte elements) from a
// row-major tensor (row stride d0_stride elements) into LDS at lds_off, with
// (pad_amount+1) DWORDs of LDS padding inserted every 8<<pad_interval bytes.
// D# layout per CDNA5 ISA ch.8.3/8.4 (group0 128b, group1 256b; 2D -> 2 groups).
__device__ __forceinline__ void tdm_load_2d(uint32_t lds_off, const void* gaddr,
                                            uint32_t tensor_d0, uint32_t tensor_d1,
                                            uint32_t tile_d0, uint32_t tile_d1,
                                            uint32_t d0_stride,
                                            uint32_t pad_interval, uint32_t pad_amount) {
  uint64_t ga = (uint64_t)(uintptr_t)gaddr;
  u32x4 g0;
  g0.x = 1u;                                  // count=1, user descriptor
  g0.y = lds_off;                             // lds_addr (bytes)
  g0.z = (uint32_t)ga;                        // global_addr[31:0]
  g0.w = (uint32_t)(ga >> 32) | (2u << 30);   // global_addr[56:32] | type=2 (image)
  u32x8 g1;
  g1.s0 = (1u << 16)                          // data_size = 1 -> 2 bytes
        | (1u << 20)                          // pad_enable
        | (pad_interval << 22) | (pad_amount << 25);
  g1.s1 = (tensor_d0 & 0xFFFFu) << 16;                                  // dim0[15:0]
  g1.s2 = ((tensor_d0 >> 16) & 0xFFFFu) | ((tensor_d1 & 0xFFFFu) << 16);// dim0 hi|dim1 lo
  g1.s3 = ((tensor_d1 >> 16) & 0xFFFFu) | (tile_d0 << 16);              // dim1 hi|tile0
  g1.s4 = tile_d1 & 0xFFFFu;                  // tile_dim1 (tile_dim2=0 -> 2D)
  g1.s5 = d0_stride;                          // tensor_dim0_stride[31:0]
  g1.s6 = 0;                                  // stride hi, dim1_stride (unused for 2D)
  g1.s7 = 0;
  asm volatile("tensor_load_to_lds %0, %1" :: "s"(g0), "s"(g1) : "memory");
}

// 16-lane all-reduce butterfly in pure VALU via DPP16.
template <int CTRL>
__device__ __forceinline__ float dpp_movf(float x) {
  return __builtin_bit_cast(
      float, __builtin_amdgcn_mov_dpp(__builtin_bit_cast(int, x), CTRL, 0xF, 0xF, true));
}
__device__ __forceinline__ float redmax16(float x) {
  x = fmaxf(x, dpp_movf<0xB1>(x));   // quad_perm [1,0,3,2]
  x = fmaxf(x, dpp_movf<0x4E>(x));   // quad_perm [2,3,0,1]
  x = fmaxf(x, dpp_movf<0x141>(x));  // row_half_mirror
  x = fmaxf(x, dpp_movf<0x140>(x));  // row_mirror
  return x;
}
__device__ __forceinline__ float redsum16(float x) {
  x += dpp_movf<0xB1>(x);
  x += dpp_movf<0x4E>(x);
  x += dpp_movf<0x141>(x);
  x += dpp_movf<0x140>(x);
  return x;
}

// Load a 16x32 bf16 A-style fragment from a row-major tile.
// lanes 0-15: row=lane, K = k0+{0..7,16..23}; lanes 16-31: row=lane-16, K=k0+{8..15,24..31}
__device__ __forceinline__ v16bf load_frag(const __bf16* base, int ld, int k0, int lane) {
  int r  = lane & 15;
  int kk = k0 + ((lane & 16) >> 1);
  const __bf16* p = base + r * ld + kk;
  union { v16bf v; uint4 q[2]; } u;
  u.q[0] = *reinterpret_cast<const uint4*>(p);
  u.q[1] = *reinterpret_cast<const uint4*>(p + 16);
  return u.v;
}

__global__ void cast_bf16_kernel(const float* __restrict__ in, __bf16* __restrict__ out, int n) {
  int i = blockIdx.x * blockDim.x + threadIdx.x;
  if (i < n) out[i] = (__bf16)in[i];
}

// W (Kd x Nd, row-major f32) -> WT (Nd x Kd, row-major bf16)
__global__ void transpose_cast_kernel(const float* __restrict__ W, __bf16* __restrict__ WT,
                                      int Kd, int Nd) {
  int i = blockIdx.x * blockDim.x + threadIdx.x;
  if (i < Kd * Nd) {
    int k = i / Nd, n = i % Nd;
    WT[(size_t)n * Kd + k] = (__bf16)W[i];
  }
}

// C(MxN) = A(MxK bf16 row-major) * W(KxN), W given as WT(NxK bf16 row-major).
// Block tile 128(M) x 64(N) x 64(K-step), 8 waves. Double-buffered LDS filled
// entirely by the Tensor Data Mover (one descriptor per tile, issued by wave 0,
// tracked with TENSORcnt); next k-tile DMA overlaps current tile's WMMAs.
// LDS rows padded 128B+16B (pad_interval=4, pad_amount=3) -> stride 72 halfwords.
template <bool SPLIT>
__global__ __launch_bounds__(256) void gemm_bf16_kernel(
    const __bf16* __restrict__ A, const __bf16* __restrict__ WT,
    float* __restrict__ outF, __bf16* __restrict__ outB,
    const float* __restrict__ bo, int M, int N, int K) {
  __shared__ __align__(16) __bf16 As[2][128 * 72];  // 128 x 64 (+8 pad)
  __shared__ __align__(16) __bf16 Bs[2][64 * 72];   // 64  x 64 (rows of W^T)

  const int tid  = threadIdx.x;
  const int w    = tid >> 5;
  const int lane = tid & 31;
  const int n0   = blockIdx.x * 64;
  const int m0   = blockIdx.y * 128;

  v8f acc[4];
#pragma unroll
  for (int t = 0; t < 4; ++t)
#pragma unroll
    for (int i = 0; i < 8; ++i) acc[t][i] = 0.f;

  auto issue_tile = [&](int bi, int k0) {  // called from wave 0 only (EXEC ignored by TDM)
    tdm_load_2d(lds_addr(&As[bi][0]), A + (size_t)m0 * K + k0,
                (uint32_t)K, (uint32_t)(M - m0), 64, 128, (uint32_t)K, 4, 3);
    tdm_load_2d(lds_addr(&Bs[bi][0]), WT + (size_t)n0 * K + k0,
                (uint32_t)K, (uint32_t)(N - n0), 64, 64, (uint32_t)K, 4, 3);
  };

  const int T = K >> 6;  // number of 64-wide k-tiles
  if (w == 0) issue_tile(0, 0);

  for (int i = 0; i < T; ++i) {
    const int cur = i & 1;
    if (w == 0) {
      if (i + 1 < T) {
        issue_tile(cur ^ 1, (i + 1) << 6);       // prefetch next tile (2 TDM ops)
        __builtin_amdgcn_s_wait_tensorcnt(2);    // current tile's 2 DMAs done
      } else {
        __builtin_amdgcn_s_wait_tensorcnt(0);
      }
    }
    __syncthreads();  // current tile visible to all waves

#pragma unroll
    for (int half = 0; half < 2; ++half) {
      const int k0 = half << 5;
      v16bf af = load_frag(&As[cur][(w * 16) * 72], 72, k0, lane);
#pragma unroll
      for (int t = 0; t < 4; ++t) {
        v16bf bfrag = load_frag(&Bs[cur][(t * 16) * 72], 72, k0, lane);
        acc[t] = wmma_bf16(af, bfrag, acc[t]);
      }
    }
    __syncthreads();  // all waves done reading before buffer is overwritten
  }

  // C layout: VGPR r -> row r (+8 for lanes>=16), col = lane&15
  const int rl = w * 16 + ((lane & 16) >> 1);
  const int cl = lane & 15;
#pragma unroll
  for (int t = 0; t < 4; ++t)
#pragma unroll
    for (int vr = 0; vr < 8; ++vr) {
      int row = m0 + rl + vr;
      int col = n0 + t * 16 + cl;
      float v = acc[t][vr];
      if (SPLIT) {
        int b = row >> 10, n = row & 1023, h = col >> 6, hd = col & 63;
        outB[((((size_t)b * HEADS + h) << 10) + n) * HDIM + hd] = (__bf16)v;
      } else {
        outF[(size_t)row * N + col] = v + bo[col];
      }
    }
}

// Flash attention with graph bias. 1 block = (b, h, 64 q rows); 4 waves x 16 rows.
__global__ __launch_bounds__(128) void attn_kernel(
    const __bf16* __restrict__ Qs, const __bf16* __restrict__ Ks,
    const __bf16* __restrict__ Vs, const float* __restrict__ bias,
    const float* __restrict__ alpha_p, __bf16* __restrict__ Obf) {
  const int tid  = threadIdx.x;
  const int w    = tid >> 5;
  const int lane = tid & 31;
  const int blk  = blockIdx.x;
  const int qt = blk & 15;
  const int h  = (blk >> 4) & (HEADS - 1);
  const int b  = blk >> 8;
  const int q0 = qt * 64;
  const float alpha = alpha_p[0];

  const size_t bh = ((size_t)b * HEADS + h) * (size_t)NSEQ * HDIM;
  const __bf16* Qbh = Qs + bh;
  const __bf16* Kbh = Ks + bh;
  const __bf16* Vbh = Vs + bh;

  __shared__ __align__(16) __bf16 Kt[64 * 72];  // keys x hd   (rows of B^T for S=Q*K^T)
  __shared__ __align__(16) __bf16 Vt[64 * 72];  // hd  x keys  (rows of B^T for O=P*V)
  __shared__ __align__(16) __bf16 Pt[64 * 72];  // q   x keys  (A for O=P*V)

  v16bf q_lo = load_frag(Qbh + (size_t)(q0 + w * 16) * HDIM, HDIM, 0, lane);
  v16bf q_hi = load_frag(Qbh + (size_t)(q0 + w * 16) * HDIM, HDIM, 32, lane);

  v8f o[4];
#pragma unroll
  for (int t = 0; t < 4; ++t)
#pragma unroll
    for (int i = 0; i < 8; ++i) o[t][i] = 0.f;
  float mrow[8], lrow[8];
#pragma unroll
  for (int i = 0; i < 8; ++i) { mrow[i] = -3.0e38f; lrow[i] = 0.f; }

  const int rl = w * 16 + ((lane & 16) >> 1);
  const int cl = lane & 15;

  for (int kb = 0; kb < NSEQ; kb += 64) {
    __syncthreads();  // all waves done reading Kt/Vt from previous block
    // K tile via async copy (ASYNCcnt), no VGPR staging
#pragma unroll
    for (int c = tid; c < 512; c += 128) {
      int r = c >> 3, kc = (c & 7) << 3;
      async_load_b128(lds_addr(&Kt[r * 72 + kc]),
                      &Kbh[(size_t)(kb + r) * HDIM + kc]);
    }
    // V^T tile: transpose on the way in (must route through VGPRs)
#pragma unroll
    for (int c = tid; c < 512; c += 128) {
      int r = c >> 3, hc = (c & 7) << 3;
      uint4 d = *reinterpret_cast<const uint4*>(&Vbh[(size_t)(kb + r) * HDIM + hc]);
      const __bf16* e = reinterpret_cast<const __bf16*>(&d);
#pragma unroll
      for (int i = 0; i < 8; ++i) Vt[(hc + i) * 72 + r] = e[i];
    }
    asm volatile("s_wait_asynccnt 0x0" ::: "memory");
    __syncthreads();

    // S = Q * K^T  (16 q-rows x 64 keys per wave)
    v8f s[4];
#pragma unroll
    for (int t = 0; t < 4; ++t) {
      v16bf k_lo = load_frag(&Kt[(t * 16) * 72], 72, 0, lane);
      v16bf k_hi = load_frag(&Kt[(t * 16) * 72], 72, 32, lane);
      v8f z;
#pragma unroll
      for (int i = 0; i < 8; ++i) z[i] = 0.f;
      z = wmma_bf16(q_lo, k_lo, z);
      s[t] = wmma_bf16(q_hi, k_hi, z);
    }

    // logits = s * SCALE + alpha * bias[q, k]
#pragma unroll
    for (int t = 0; t < 4; ++t)
#pragma unroll
      for (int vr = 0; vr < 8; ++vr) {
        int qr = q0 + rl + vr;
        int kc = kb + t * 16 + cl;
        s[t][vr] = s[t][vr] * 0.125f + alpha * bias[(size_t)qr * NSEQ + kc];
      }

    // online softmax per q-row; row values live in one 16-lane half -> DPP butterfly
#pragma unroll
    for (int vr = 0; vr < 8; ++vr) {
      float mx = fmaxf(fmaxf(s[0][vr], s[1][vr]), fmaxf(s[2][vr], s[3][vr]));
      mx = redmax16(mx);
      float mnew = fmaxf(mrow[vr], mx);
      float corr = __expf(mrow[vr] - mnew);
      float ssum = 0.f;
#pragma unroll
      for (int t = 0; t < 4; ++t) {
        float p = __expf(s[t][vr] - mnew);
        s[t][vr] = p;
        ssum += p;
      }
      ssum = redsum16(ssum);
      lrow[vr] = lrow[vr] * corr + ssum;
      mrow[vr] = mnew;
#pragma unroll
      for (int t = 0; t < 4; ++t) o[t][vr] *= corr;
    }

    // P -> LDS (each wave writes & reads only its own 16 rows; LDS in-order per wave)
#pragma unroll
    for (int t = 0; t < 4; ++t)
#pragma unroll
      for (int vr = 0; vr < 8; ++vr)
        Pt[(rl + vr) * 72 + t * 16 + cl] = (__bf16)s[t][vr];

    v16bf p_lo = load_frag(&Pt[(w * 16) * 72], 72, 0, lane);
    v16bf p_hi = load_frag(&Pt[(w * 16) * 72], 72, 32, lane);
#pragma unroll
    for (int t = 0; t < 4; ++t) {
      v16bf v_lo = load_frag(&Vt[(t * 16) * 72], 72, 0, lane);
      v16bf v_hi = load_frag(&Vt[(t * 16) * 72], 72, 32, lane);
      o[t] = wmma_bf16(p_lo, v_lo, o[t]);
      o[t] = wmma_bf16(p_hi, v_hi, o[t]);
    }
  }

  // normalize and write merged-head bf16 output [B, NSEQ, DIM]
#pragma unroll
  for (int t = 0; t < 4; ++t)
#pragma unroll
    for (int vr = 0; vr < 8; ++vr) {
      float val = o[t][vr] / lrow[vr];
      int qr = q0 + rl + vr;
      int hd = t * 16 + cl;
      Obf[((size_t)b * NSEQ + qr) * DIM + (size_t)h * HDIM + hd] = (__bf16)val;
    }
}

extern "C" void kernel_launch(void* const* d_in, const int* in_sizes, int n_in,
                              void* d_out, int out_size, void* d_ws, size_t ws_size,
                              hipStream_t stream) {
  (void)in_sizes; (void)n_in; (void)out_size; (void)ws_size;
  const float* query = (const float*)d_in[0];
  const float* value = (const float*)d_in[1];
  const float* bias  = (const float*)d_in[2];
  const float* Wq    = (const float*)d_in[3];
  const float* Wk    = (const float*)d_in[4];
  const float* Wv    = (const float*)d_in[5];
  const float* Wo    = (const float*)d_in[6];
  const float* bo    = (const float*)d_in[7];
  const float* alpha = (const float*)d_in[8];
  float* out = (float*)d_out;

  const size_t nBND = (size_t)B_ * NSEQ * DIM;  // 4M elems
  const size_t nDD  = (size_t)DIM * DIM;        // 1M elems
  char* ws = (char*)d_ws;
  size_t off = 0;
  auto take = [&](size_t elems) {
    __bf16* p = (__bf16*)(ws + off);
    off += elems * sizeof(__bf16);
    return p;
  };
  __bf16* qbf = take(nBND);
  __bf16* vbf = take(nBND);
  __bf16* WqT = take(nDD);
  __bf16* WkT = take(nDD);
  __bf16* WvT = take(nDD);
  __bf16* WoT = take(nDD);
  __bf16* Qsp = take(nBND);
  __bf16* Ksp = take(nBND);
  __bf16* Vsp = take(nBND);
  __bf16* Obf = take(nBND);  // total ~56 MB of workspace

  cast_bf16_kernel<<<(int)(nBND / 256), 256, 0, stream>>>(query, qbf, (int)nBND);
  cast_bf16_kernel<<<(int)(nBND / 256), 256, 0, stream>>>(value, vbf, (int)nBND);
  transpose_cast_kernel<<<(int)(nDD / 256), 256, 0, stream>>>(Wq, WqT, DIM, DIM);
  transpose_cast_kernel<<<(int)(nDD / 256), 256, 0, stream>>>(Wk, WkT, DIM, DIM);
  transpose_cast_kernel<<<(int)(nDD / 256), 256, 0, stream>>>(Wv, WvT, DIM, DIM);
  transpose_cast_kernel<<<(int)(nDD / 256), 256, 0, stream>>>(Wo, WoT, DIM, DIM);

  dim3 g(DIM / 64, (B_ * NSEQ) / 128);
  gemm_bf16_kernel<true><<<g, 256, 0, stream>>>(qbf, WqT, nullptr, Qsp, nullptr,
                                                B_ * NSEQ, DIM, DIM);
  gemm_bf16_kernel<true><<<g, 256, 0, stream>>>(vbf, WkT, nullptr, Ksp, nullptr,
                                                B_ * NSEQ, DIM, DIM);
  gemm_bf16_kernel<true><<<g, 256, 0, stream>>>(vbf, WvT, nullptr, Vsp, nullptr,
                                                B_ * NSEQ, DIM, DIM);

  attn_kernel<<<B_ * HEADS * (NSEQ / 64), 128, 0, stream>>>(Qsp, Ksp, Vsp, bias, alpha, Obf);

  gemm_bf16_kernel<false><<<g, 256, 0, stream>>>(Obf, WoT, out, nullptr, bo,
                                                 B_ * NSEQ, DIM, DIM);
}
